// AEM_71459665871084
// MI455X (gfx1250) — matrix operand order repaired and checked by reading, not compiled, not run on hardware
//
#include <hip/hip_runtime.h>
#include <math.h>

typedef __bf16 bf16;
typedef __attribute__((ext_vector_type(16))) __bf16 v16bf;
typedef __attribute__((ext_vector_type(8)))  float  v8f;
typedef __attribute__((ext_vector_type(4)))  float  v4f;
typedef __attribute__((ext_vector_type(4)))  unsigned int uint4v;
typedef __attribute__((ext_vector_type(2)))  unsigned int uint2v;

#define B_SZ 1024
#define U_SZ 32
#define D_SZ 1024
#define SA   40            // padded LDS row stride in halfs -> conflict-free b128 frag reads
#define TILE_HALFS (64 * SA)          // one operand tile
#define BUF_HALFS  (4 * TILE_HALFS)   // Ahi|Alo|Bhi|Blo

union FragAB { v16bf v; uint4v u[2]; };
union Pack4  { bf16 h[4]; uint2v u; };

// ---------------------------------------------------------------------------
// zero scratch
// ---------------------------------------------------------------------------
__global__ void aem_zero_f32(float* p, int n) {
  int i = blockIdx.x * 256 + threadIdx.x;
  if (i < n) p[i] = 0.f;
}

// ---------------------------------------------------------------------------
// shared inner machinery: fragment gather + 6 compensated WMMAs on one buffer
// ---------------------------------------------------------------------------
__device__ __forceinline__
void wmma_step(const bf16* Ahi, const bf16* Alo, const bf16* Bhi, const bf16* Blo,
               int lane, int mw, int nw, v8f& c0, v8f& c1)
{
  const int m  = mw * 16 + (lane & 15);
  const int qa = (lane < 16) ? 0 : 16;
  FragAB ahi, alo;
  ahi.u[0] = *(const uint4v*)(Ahi + m * SA + qa);
  ahi.u[1] = *(const uint4v*)(Ahi + m * SA + qa + 8);
  alo.u[0] = *(const uint4v*)(Alo + m * SA + qa);
  alo.u[1] = *(const uint4v*)(Alo + m * SA + qa + 8);

  const int qb  = (lane < 16) ? 0 : 16;
  const int dc0 = nw * 32 + (lane & 15);
  const int dc1 = dc0 + 16;
  FragAB bh0, bl0, bh1, bl1;
  bh0.u[0] = *(const uint4v*)(Bhi + dc0 * SA + qb);
  bh0.u[1] = *(const uint4v*)(Bhi + dc0 * SA + qb + 8);
  bl0.u[0] = *(const uint4v*)(Blo + dc0 * SA + qb);
  bl0.u[1] = *(const uint4v*)(Blo + dc0 * SA + qb + 8);
  bh1.u[0] = *(const uint4v*)(Bhi + dc1 * SA + qb);
  bh1.u[1] = *(const uint4v*)(Bhi + dc1 * SA + qb + 8);
  bl1.u[0] = *(const uint4v*)(Blo + dc1 * SA + qb);
  bl1.u[1] = *(const uint4v*)(Blo + dc1 * SA + qb + 8);

  c0 = __builtin_amdgcn_wmma_f32_16x16x32_bf16(false, ahi.v, false, bh0.v, (short)0, c0, false, false);
  c0 = __builtin_amdgcn_wmma_f32_16x16x32_bf16(false, ahi.v, false, bl0.v, (short)0, c0, false, false);
  c0 = __builtin_amdgcn_wmma_f32_16x16x32_bf16(false, alo.v, false, bh0.v, (short)0, c0, false, false);
  c1 = __builtin_amdgcn_wmma_f32_16x16x32_bf16(false, ahi.v, false, bh1.v, (short)0, c1, false, false);
  c1 = __builtin_amdgcn_wmma_f32_16x16x32_bf16(false, ahi.v, false, bl1.v, (short)0, c1, false, false);
  c1 = __builtin_amdgcn_wmma_f32_16x16x32_bf16(false, alo.v, false, bh1.v, (short)0, c1, false, false);
}

__device__ __forceinline__
void split_store4(bf16* hiDst, bf16* loDst, v4f x)
{
  Pack4 ph, pl;
#pragma unroll
  for (int j = 0; j < 4; ++j) {
    bf16 h = (bf16)x[j];
    ph.h[j] = h;
    pl.h[j] = (bf16)(x[j] - (float)h);
  }
  *(uint2v*)hiDst = ph.u;
  *(uint2v*)loDst = pl.u;
}

// ---------------------------------------------------------------------------
// Kernel 2: k = input_h @ key_weight^T + bias ; store bf16 hi/lo split
// Double-buffered LDS pipeline.
// ---------------------------------------------------------------------------
__global__ __launch_bounds__(256)
void aem_key_proj(const float* __restrict__ X,    // [B, D]
                  const float* __restrict__ KW,   // [D, D]
                  const float* __restrict__ bias, // [D]
                  bf16* __restrict__ khi, bf16* __restrict__ klo)
{
  __shared__ __align__(16) char smem[2 * BUF_HALFS * 2];  // 40 KB

  const int b0 = blockIdx.y * 64;
  const int d0 = blockIdx.x * 64;
  const int t    = threadIdx.x;
  const int lane = t & 31;
  const int wid  = t >> 5;
  const int mw = wid & 3;
  const int nw = wid >> 2;

  const int arow = t >> 2;        // 0..63
  const int acol = (t & 3) * 8;   // 0,8,16,24
  const int wrow = t >> 3;        // 0..31
  const int wcol = (t & 7) * 4;   // 0..28

  v8f c0 = {}; v8f c1 = {};

  // staging registers
  v4f sA0, sA1, sW0, sW1;

  const float* Aptr  = X  + (size_t)(b0 + arow) * D_SZ + acol;
  const float* Wp0   = KW + (size_t)(d0 + wrow) * D_SZ + wcol;
  const float* Wp1   = KW + (size_t)(d0 + wrow + 32) * D_SZ + wcol;

  // prologue: stage k-step 0
  sA0 = *(const v4f*)(Aptr);
  sA1 = *(const v4f*)(Aptr + 4);
  sW0 = *(const v4f*)(Wp0);
  sW1 = *(const v4f*)(Wp1);
  {
    bf16* Ahi = (bf16*)smem;
    split_store4(Ahi + arow * SA + acol,     Ahi + TILE_HALFS + arow * SA + acol,     sA0);
    split_store4(Ahi + arow * SA + acol + 4, Ahi + TILE_HALFS + arow * SA + acol + 4, sA1);
    split_store4(Ahi + 2 * TILE_HALFS + wrow * SA + wcol,        Ahi + 3 * TILE_HALFS + wrow * SA + wcol,        sW0);
    split_store4(Ahi + 2 * TILE_HALFS + (wrow + 32) * SA + wcol, Ahi + 3 * TILE_HALFS + (wrow + 32) * SA + wcol, sW1);
  }
  __syncthreads();

  for (int step = 0; step < D_SZ / 32; ++step) {
    bf16* cur = (bf16*)smem + (step & 1) * BUF_HALFS;
    bf16* nxt = (bf16*)smem + ((step + 1) & 1) * BUF_HALFS;
    const bool more = (step + 1) < D_SZ / 32;

    if (more) {                       // issue next tile's global loads first
      int kk = (step + 1) * 32;
      sA0 = *(const v4f*)(Aptr + kk);
      sA1 = *(const v4f*)(Aptr + kk + 4);
      sW0 = *(const v4f*)(Wp0 + kk);
      sW1 = *(const v4f*)(Wp1 + kk);
    }

    wmma_step(cur, cur + TILE_HALFS, cur + 2 * TILE_HALFS, cur + 3 * TILE_HALFS,
              lane, mw, nw, c0, c1);

    if (more) {                       // convert + park next tile
      split_store4(nxt + arow * SA + acol,     nxt + TILE_HALFS + arow * SA + acol,     sA0);
      split_store4(nxt + arow * SA + acol + 4, nxt + TILE_HALFS + arow * SA + acol + 4, sA1);
      split_store4(nxt + 2 * TILE_HALFS + wrow * SA + wcol,        nxt + 3 * TILE_HALFS + wrow * SA + wcol,        sW0);
      split_store4(nxt + 2 * TILE_HALFS + (wrow + 32) * SA + wcol, nxt + 3 * TILE_HALFS + (wrow + 32) * SA + wcol, sW1);
    }
    __syncthreads();
  }

  // epilogue: add bias, store split-bf16 k
  const int crow0 = mw * 16 + ((lane < 16) ? 0 : 8);
  const int ccol0 = nw * 32 + (lane & 15);
  const float bias0 = bias[d0 + ccol0];
  const float bias1 = bias[d0 + ccol0 + 16];
#pragma unroll
  for (int v = 0; v < 8; ++v) {
    size_t i0 = (size_t)(b0 + crow0 + v) * D_SZ + d0 + ccol0;
    float k0 = c0[v] + bias0;
    float k1 = c1[v] + bias1;
    bf16 h0 = (bf16)k0, h1 = (bf16)k1;
    khi[i0]      = h0; klo[i0]      = (bf16)(k0 - (float)h0);
    khi[i0 + 16] = h1; klo[i0 + 16] = (bf16)(k1 - (float)h1);
  }
}

// ---------------------------------------------------------------------------
// Kernel 3: per-unit GEMM kk_u = k @ W_u^T, fused dot with dah -> raw scores
// Double-buffered LDS pipeline.
// ---------------------------------------------------------------------------
__global__ __launch_bounds__(256)
void aem_unit_score(const bf16* __restrict__ khi, const bf16* __restrict__ klo,
                    const float* __restrict__ W,    // [U, D, D]
                    const float* __restrict__ dah,  // [B, U, D]
                    float* __restrict__ raw)        // [B, U]
{
  __shared__ __align__(16) char smem[2 * BUF_HALFS * 2];  // 40 KB

  const int u  = blockIdx.z;
  const int b0 = blockIdx.y * 64;
  const int d0 = blockIdx.x * 64;
  const int t    = threadIdx.x;
  const int lane = t & 31;
  const int wid  = t >> 5;
  const int mw = wid & 3;
  const int nw = wid >> 2;

  const int arow = t >> 2;
  const int acol = (t & 3) * 8;
  const int wrow = t >> 3;
  const int wcol = (t & 7) * 4;

  v8f c0 = {}; v8f c1 = {};

  uint4v sAh, sAl;   // pre-split bf16 k rows
  v4f    sW0, sW1;   // f32 W rows

  const bf16*  Ahp = khi + (size_t)(b0 + arow) * D_SZ + acol;
  const bf16*  Alp = klo + (size_t)(b0 + arow) * D_SZ + acol;
  const float* Wp0 = W + ((size_t)u * D_SZ + d0 + wrow) * D_SZ + wcol;
  const float* Wp1 = W + ((size_t)u * D_SZ + d0 + wrow + 32) * D_SZ + wcol;

  // prologue: stage k-step 0
  sAh = *(const uint4v*)(Ahp);
  sAl = *(const uint4v*)(Alp);
  sW0 = *(const v4f*)(Wp0);
  sW1 = *(const v4f*)(Wp1);
  {
    bf16* buf = (bf16*)smem;
    *(uint4v*)(buf + arow * SA + acol)              = sAh;
    *(uint4v*)(buf + TILE_HALFS + arow * SA + acol) = sAl;
    split_store4(buf + 2 * TILE_HALFS + wrow * SA + wcol,        buf + 3 * TILE_HALFS + wrow * SA + wcol,        sW0);
    split_store4(buf + 2 * TILE_HALFS + (wrow + 32) * SA + wcol, buf + 3 * TILE_HALFS + (wrow + 32) * SA + wcol, sW1);
  }
  __syncthreads();

  for (int step = 0; step < D_SZ / 32; ++step) {
    bf16* cur = (bf16*)smem + (step & 1) * BUF_HALFS;
    bf16* nxt = (bf16*)smem + ((step + 1) & 1) * BUF_HALFS;
    const bool more = (step + 1) < D_SZ / 32;

    if (more) {
      int kk = (step + 1) * 32;
      sAh = *(const uint4v*)(Ahp + kk);
      sAl = *(const uint4v*)(Alp + kk);
      sW0 = *(const v4f*)(Wp0 + kk);
      sW1 = *(const v4f*)(Wp1 + kk);
    }

    wmma_step(cur, cur + TILE_HALFS, cur + 2 * TILE_HALFS, cur + 3 * TILE_HALFS,
              lane, mw, nw, c0, c1);

    if (more) {
      *(uint4v*)(nxt + arow * SA + acol)              = sAh;
      *(uint4v*)(nxt + TILE_HALFS + arow * SA + acol) = sAl;
      split_store4(nxt + 2 * TILE_HALFS + wrow * SA + wcol,        nxt + 3 * TILE_HALFS + wrow * SA + wcol,        sW0);
      split_store4(nxt + 2 * TILE_HALFS + (wrow + 32) * SA + wcol, nxt + 3 * TILE_HALFS + (wrow + 32) * SA + wcol, sW1);
    }
    __syncthreads();
  }

  // park C tile in LDS, then fused coalesced dot with dah and reduce
  float* Csh = (float*)smem;  // [64][68] f32 (17.4 KB, fits)
  const int crow0 = mw * 16 + ((lane < 16) ? 0 : 8);
  const int ccol0 = nw * 32 + (lane & 15);
#pragma unroll
  for (int v = 0; v < 8; ++v) {
    Csh[(crow0 + v) * 68 + ccol0]      = c0[v];
    Csh[(crow0 + v) * 68 + ccol0 + 16] = c1[v];
  }
  __syncthreads();

  const int rrow = t >> 2;          // batch row within tile
  const int rseg = (t & 3) * 16;    // 16-col segment
  const float* dptr = dah + ((size_t)(b0 + rrow) * U_SZ + u) * D_SZ + d0 + rseg;
  float s = 0.f;
#pragma unroll
  for (int j = 0; j < 16; j += 4) {
    v4f xv = *(const v4f*)(dptr + j);
    s += xv[0] * Csh[rrow * 68 + rseg + j]
       + xv[1] * Csh[rrow * 68 + rseg + j + 1]
       + xv[2] * Csh[rrow * 68 + rseg + j + 2]
       + xv[3] * Csh[rrow * 68 + rseg + j + 3];
  }
  s += __shfl_down(s, 2);
  s += __shfl_down(s, 1);
  if ((t & 3) == 0)
    atomicAdd(&raw[(size_t)(b0 + rrow) * U_SZ + u], s);
}

// ---------------------------------------------------------------------------
// Kernel 4: sigmoid, mask(u<=t), normalize -> scores + weights
// ---------------------------------------------------------------------------
__global__ __launch_bounds__(32)
void aem_normalize(const float* __restrict__ raw, const int* __restrict__ tptr,
                   float* __restrict__ scores_out, float* __restrict__ wbuf)
{
  const int b = blockIdx.x;
  const int u = threadIdx.x;
  const int tv = *tptr;
  float x = raw[b * U_SZ + u] * 0.03125f;  // 1/sqrt(1024)
  float s = 1.f / (1.f + __expf(-x));
  if (u > tv) s = 0.f;
  float tot = s;
#pragma unroll
  for (int off = 16; off > 0; off >>= 1) tot += __shfl_xor(tot, off);
  scores_out[b * U_SZ + u] = s;
  wbuf[b * U_SZ + u] = s / tot;
}

// ---------------------------------------------------------------------------
// Kernel 5: out[b,:] = sum_u w[b,u] * dah[b,u,:]  (skip masked units)
// ---------------------------------------------------------------------------
__global__ __launch_bounds__(256)
void aem_pool(const float* __restrict__ dah, const float* __restrict__ wbuf,
              float* __restrict__ out)
{
  __shared__ float wl[U_SZ];
  const int b = blockIdx.y;
  const int d = blockIdx.x * 256 + threadIdx.x;
  if (threadIdx.x < U_SZ) wl[threadIdx.x] = wbuf[b * U_SZ + threadIdx.x];
  __syncthreads();
  float acc = 0.f;
#pragma unroll
  for (int u = 0; u < U_SZ; ++u) {
    float wu = wl[u];                 // uniform across block -> scalar branch
    if (wu != 0.f)
      acc += wu * dah[((size_t)b * U_SZ + u) * D_SZ + d];
  }
  out[(size_t)b * D_SZ + d] = acc;
}

// ---------------------------------------------------------------------------
extern "C" void kernel_launch(void* const* d_in, const int* in_sizes, int n_in,
                              void* d_out, int out_size, void* d_ws, size_t ws_size,
                              hipStream_t stream)
{
  const float* dah  = (const float*)d_in[0];  // [B, U, D]
  const float* X    = (const float*)d_in[1];  // [B, D]
  const int*   tptr = (const int*)d_in[2];    // scalar
  const float* W    = (const float*)d_in[3];  // [U, D, D]
  const float* KW   = (const float*)d_in[4];  // [D, D]
  const float* bias = (const float*)d_in[5];  // [D]

  float* out    = (float*)d_out;               // [B, D]
  float* scores = out + (size_t)B_SZ * D_SZ;   // [B, U]

  char* ws = (char*)d_ws;
  bf16*  khi  = (bf16*)ws;                                   // 2 MB
  bf16*  klo  = (bf16*)(ws + (size_t)B_SZ * D_SZ * 2);       // 2 MB
  float* raw  = (float*)(ws + (size_t)B_SZ * D_SZ * 4);      // 128 KB
  float* wbuf = raw + (size_t)B_SZ * U_SZ;                   // 128 KB

  aem_zero_f32<<<(B_SZ * U_SZ + 255) / 256, 256, 0, stream>>>(raw, B_SZ * U_SZ);
  aem_key_proj<<<dim3(D_SZ / 64, B_SZ / 64), 256, 0, stream>>>(X, KW, bias, khi, klo);
  aem_unit_score<<<dim3(D_SZ / 64, B_SZ / 64, U_SZ), 256, 0, stream>>>(khi, klo, W, dah, raw);
  aem_normalize<<<B_SZ, 32, 0, stream>>>(raw, tptr, scores, wbuf);
  aem_pool<<<dim3(D_SZ / 256, B_SZ), 256, 0, stream>>>(dah, wbuf, out);
}
